// VoxelBackBone8x_Decoder_59030030516563
// MI455X (gfx1250) — matrix operand order repaired and checked by reading, not compiled, not run on hardware
//
#include <hip/hip_runtime.h>

typedef __attribute__((ext_vector_type(16))) __bf16 bf16x16;
typedef __attribute__((ext_vector_type(8)))  float  f32x8;

__device__ __forceinline__ unsigned short f2bf(float f) {
    unsigned u = __float_as_uint(f);
    unsigned r = u + 0x7FFFu + ((u >> 16) & 1u);   // round to nearest even
    return (unsigned short)(r >> 16);
}

// ---------------------------------------------------------------------------
// fp32 -> bf16 feature conversion
// ---------------------------------------------------------------------------
__global__ void f32_to_bf16_kernel(const float* __restrict__ in,
                                   unsigned short* __restrict__ out, long n) {
    long i = (long)blockIdx.x * blockDim.x + threadIdx.x;
    if (i < n) out[i] = f2bf(in[i]);
}

// ---------------------------------------------------------------------------
// Pack W (27, ci, co) fp32 into lane-ordered bf16 B-fragments:
//   wpack[((nt*KC + kc)*32 + lane)*16 + e],
//   K = kc*32 + (lane<16?0:8) + (e&7) + 16*(e>>3),  N = (lane&15) + nt*16
// Zero-padded for K >= 27*ci or N >= co.
// ---------------------------------------------------------------------------
__global__ void pack_w_kernel(const float* __restrict__ W,
                              unsigned short* __restrict__ wpack,
                              int ci, int cishift, int co, int KC, int cotiles) {
    int idx = blockIdx.x * blockDim.x + threadIdx.x;
    int total = cotiles * KC * 512;
    if (idx >= total) return;
    int e    = idx & 15;
    int lane = (idx >> 4) & 31;
    int kc   = (idx >> 9) % KC;
    int nt   = idx / (KC * 512);
    int K = kc * 32 + ((lane >> 4) << 3) + (e & 7) + ((e >> 3) << 4);
    int N = (lane & 15) + nt * 16;
    int KTOT = 27 << cishift;
    unsigned short v = 0;
    if (K < KTOT && N < co) {
        int tap = K >> cishift;
        int kin = K & (ci - 1);
        v = f2bf(W[((long)tap * ci + kin) * co + N]);
    }
    wpack[idx] = v;
}

// ---------------------------------------------------------------------------
// Gather-GEMM. One wave owns TWO 16-row M-tiles and COTILES 16-col tiles.
//   - fully branchless gathers (index clamp + address cndmask, no exec ops)
//   - tap-outer loop (ci>=32): one nbr row load per tap, prefetched one tap
//     ahead so the row->feature dependent chain hides under 8-16 WMMAs
//   - A-fragment data double-buffered one K-chunk ahead
//   - all B fragments for a chunk loaded to distinct regs (single clause)
// 128 threads = 4 waves = 128 rows per workgroup.
// ---------------------------------------------------------------------------
template <int COTILES, int CISHIFT>
__global__ __launch_bounds__(128) void gemm_kernel(
    const unsigned short* __restrict__ fin,   // (Nin, CI) bf16
    const int* __restrict__ nbr,              // (Nout, 27)
    const unsigned short* __restrict__ wpack, // packed B fragments
    const unsigned short* __restrict__ zp,    // 64B zero page
    float* __restrict__ y,                    // (Nout, COPAD) fp32 pre-BN
    float* __restrict__ stats,                // [0..63]=sum  [64..127]=sumsq
    int Nout) {
    constexpr int CI    = 1 << CISHIFT;
    constexpr int KTOT  = 27 * CI;
    constexpr int KC    = (KTOT + 31) / 32;
    constexpr int COPAD = COTILES * 16;

    const int lane   = threadIdx.x & 31;
    const int wave   = threadIdx.x >> 5;
    const int mBase0 = (blockIdx.x * 4 + wave) * 32;   // first of 2 row tiles
    const int M      = lane & 15;
    const int hi     = lane >> 4;
    const int baseA  = hi * 8;

    const int n0 = mBase0 + M;
    const int n1 = n0 + 16;
    const bool v0 = n0 < Nout;
    const bool v1 = n1 < Nout;
    const int nb0 = n0 * 27;
    const int nb1 = n1 * 27;

    union AB { uint4 q[2]; bf16x16 v; };

    // Branchless neighbor-row load: clamp index, fix up result.
    auto ldRow = [&](int nb, bool valid, int tap) -> int {
        int off = valid ? (nb + tap) : 0;
        int row = nbr[off];
        return valid ? row : -1;
    };
    // Branchless A-fragment data load (two contiguous b128 from one row).
    auto ldAData = [&](int row, int kin) -> AB {
        AB a;
        const unsigned short* p =
            (row >= 0) ? (fin + (size_t)row * CI + kin) : zp;
        a.q[0] = *(const uint4*)p;
        a.q[1] = *(const uint4*)(p + 16);
        return a;
    };

    f32x8 acc[2][COTILES];
#pragma unroll
    for (int j = 0; j < 2; ++j)
#pragma unroll
        for (int t = 0; t < COTILES; ++t) acc[j][t] = (f32x8)0.0f;

    // -------- matrix-multiply main loops --------
    if constexpr (CISHIFT >= 5) {
        constexpr int CPT = CI / 32;   // K-chunks per tap (2 for ci=64, 1 for ci=32)

        int rowCur0 = ldRow(nb0, v0, 0);
        int rowCur1 = ldRow(nb1, v1, 0);
        AB aCur0 = ldAData(rowCur0, baseA);
        AB aCur1 = ldAData(rowCur1, baseA);

        for (int tap = 0; tap < 27; ++tap) {
            // prefetch next tap's rows (consumed near end of this iteration)
            const int tapN = (tap + 1 < 27) ? (tap + 1) : tap;
            int rowNxt0 = ldRow(nb0, v0, tapN);
            int rowNxt1 = ldRow(nb1, v1, tapN);

#pragma unroll
            for (int c = 0; c < CPT; ++c) {
                const int kc = tap * CPT + c;

                // B fragments for this chunk: distinct regs, one clause
                uint4 bq[COTILES][2];
#pragma unroll
                for (int nt = 0; nt < COTILES; ++nt) {
                    const uint4* wp = (const uint4*)(
                        wpack + (((nt * KC + kc) * 32 + lane) << 4));
                    bq[nt][0] = wp[0];
                    bq[nt][1] = wp[1];
                }

                // prefetch A data for the next chunk (overlaps WMMAs)
                const bool lastC = (c == CPT - 1);
                const int prow0 = lastC ? rowNxt0 : rowCur0;
                const int prow1 = lastC ? rowNxt1 : rowCur1;
                const int kinN  = lastC ? baseA : (c + 1) * 32 + baseA;
                AB aNxt0 = ldAData(prow0, kinN);
                AB aNxt1 = ldAData(prow1, kinN);

                // 2 * COTILES back-to-back WMMAs
#pragma unroll
                for (int nt = 0; nt < COTILES; ++nt) {
                    AB b;
                    b.q[0] = bq[nt][0];
                    b.q[1] = bq[nt][1];
                    acc[0][nt] = __builtin_amdgcn_wmma_f32_16x16x32_bf16(
                        false, aCur0.v, false, b.v, (short)0, acc[0][nt],
                        false, false);
                    acc[1][nt] = __builtin_amdgcn_wmma_f32_16x16x32_bf16(
                        false, aCur1.v, false, b.v, (short)0, acc[1][nt],
                        false, false);
                }

                aCur0 = aNxt0;
                aCur1 = aNxt1;
            }
            rowCur0 = rowNxt0;
            rowCur1 = rowNxt1;
        }
    } else {
        // ci == 16: chunk spans taps 2t and 2t+1; per-chunk branchless rows
        auto loadA16 = [&](int nb, bool valid, int kc) -> AB {
            AB a;
            int gkA = kc * 32 + baseA;
            int gkB = gkA + 16;
            int rowA = ldRow(nb, valid && (gkA < KTOT), gkA >> CISHIFT);
            int rowB = ldRow(nb, valid && (gkB < KTOT), gkB >> CISHIFT);
            const unsigned short* pA =
                (rowA >= 0) ? (fin + (size_t)rowA * CI + (gkA & (CI - 1))) : zp;
            const unsigned short* pB =
                (rowB >= 0) ? (fin + (size_t)rowB * CI + (gkB & (CI - 1))) : zp;
            a.q[0] = *(const uint4*)pA;
            a.q[1] = *(const uint4*)pB;
            return a;
        };

        AB aCur0 = loadA16(nb0, v0, 0);
        AB aCur1 = loadA16(nb1, v1, 0);

        for (int kc = 0; kc < KC; ++kc) {
            uint4 bq[COTILES][2];
#pragma unroll
            for (int nt = 0; nt < COTILES; ++nt) {
                const uint4* wp = (const uint4*)(
                    wpack + (((nt * KC + kc) * 32 + lane) << 4));
                bq[nt][0] = wp[0];
                bq[nt][1] = wp[1];
            }

            AB aNxt0, aNxt1;
            if (kc + 1 < KC) {
                aNxt0 = loadA16(nb0, v0, kc + 1);
                aNxt1 = loadA16(nb1, v1, kc + 1);
            } else {
                aNxt0 = aCur0;
                aNxt1 = aCur1;
            }

#pragma unroll
            for (int nt = 0; nt < COTILES; ++nt) {
                AB b;
                b.q[0] = bq[nt][0];
                b.q[1] = bq[nt][1];
                acc[0][nt] = __builtin_amdgcn_wmma_f32_16x16x32_bf16(
                    false, aCur0.v, false, b.v, (short)0, acc[0][nt],
                    false, false);
                acc[1][nt] = __builtin_amdgcn_wmma_f32_16x16x32_bf16(
                    false, aCur1.v, false, b.v, (short)0, acc[1][nt],
                    false, false);
            }

            aCur0 = aNxt0;
            aCur1 = aNxt1;
        }
    }

    // Store pre-BN y and accumulate BN statistics.
    // C/D layout: VGPR r -> row tileBase + hi*8 + r, col = nt*16 + (lane&15)
#pragma unroll
    for (int j = 0; j < 2; ++j) {
        const int tileBase = mBase0 + j * 16;
#pragma unroll
        for (int nt = 0; nt < COTILES; ++nt) {
            const int col = nt * 16 + M;
            float s = 0.f, sq = 0.f;
#pragma unroll
            for (int r = 0; r < 8; ++r) {
                int row = tileBase + hi * 8 + r;
                float v = acc[j][nt][r];
                if (row < Nout) {
                    y[(long)row * COPAD + col] = v;
                    s  += v;
                    sq += v * v;
                }
            }
            atomicAdd(&stats[col], s);
            atomicAdd(&stats[64 + col], sq);
        }
    }
}

// ---------------------------------------------------------------------------
// BN finalize: mu, rsqrt(var + 1e-3)
// ---------------------------------------------------------------------------
__global__ void bn_finalize_kernel(float* __restrict__ stats, int Nout, int co) {
    int c = threadIdx.x;
    if (c < co) {
        float inv = 1.0f / (float)Nout;
        float mu  = stats[c] * inv;
        float var = stats[64 + c] * inv - mu * mu;
        stats[128 + c] = mu;
        stats[192 + c] = rsqrtf(var + 1e-3f);
    }
}

// ---------------------------------------------------------------------------
// BN apply (+optional ReLU); writes bf16 features for next layer, or fp32
// final output.
// ---------------------------------------------------------------------------
template <bool RELU, bool FINAL>
__global__ void bn_apply_kernel(const float* __restrict__ y,
                                const float* __restrict__ stats,
                                const float* __restrict__ g,
                                const float* __restrict__ b,
                                unsigned short* __restrict__ outbf,
                                float* __restrict__ outf,
                                int Nout, int co, int copad) {
    long i = (long)blockIdx.x * blockDim.x + threadIdx.x;
    long total = (long)Nout * co;
    if (i >= total) return;
    int  c = (int)(i % co);
    long n = i / co;
    float t = (y[n * copad + c] - stats[128 + c]) * stats[192 + c] * g[c] + b[c];
    if (RELU) t = fmaxf(t, 0.0f);
    if (FINAL) outf[i] = t;
    else       outbf[i] = f2bf(t);
}

// ---------------------------------------------------------------------------
// Per-block driver
// ---------------------------------------------------------------------------
template <int COTILES, int CISHIFT>
static void run_block(const unsigned short* fin, const int* nbr, const float* W,
                      const float* g, const float* b, int Nout, int co,
                      unsigned short* fout, float* foutF, bool relu, bool fin_,
                      float* yBuf, unsigned short* wpack,
                      const unsigned short* zp, float* stats,
                      hipStream_t stream) {
    constexpr int CI = 1 << CISHIFT;
    constexpr int KC = (27 * CI + 31) / 32;
    int packTotal = COTILES * KC * 512;
    pack_w_kernel<<<(packTotal + 255) / 256, 256, 0, stream>>>(
        W, wpack, CI, CISHIFT, co, KC, COTILES);
    hipMemsetAsync(stats, 0, 512, stream);
    int grid = (Nout + 127) / 128;
    gemm_kernel<COTILES, CISHIFT><<<grid, 128, 0, stream>>>(
        fin, nbr, wpack, zp, yBuf, stats, Nout);
    bn_finalize_kernel<<<1, 64, 0, stream>>>(stats, Nout, co);
    long total = (long)Nout * co;
    int ap = (int)((total + 255) / 256);
    if (fin_)
        bn_apply_kernel<false, true><<<ap, 256, 0, stream>>>(
            yBuf, stats, g, b, nullptr, foutF, Nout, co, COTILES * 16);
    else if (relu)
        bn_apply_kernel<true, false><<<ap, 256, 0, stream>>>(
            yBuf, stats, g, b, fout, nullptr, Nout, co, COTILES * 16);
    else
        bn_apply_kernel<false, false><<<ap, 256, 0, stream>>>(
            yBuf, stats, g, b, fout, nullptr, Nout, co, COTILES * 16);
}

extern "C" void kernel_launch(void* const* d_in, const int* in_sizes, int n_in,
                              void* d_out, int out_size, void* d_ws, size_t ws_size,
                              hipStream_t stream) {
    (void)n_in; (void)out_size; (void)ws_size;

    const int n4 = in_sizes[0] / 27;
    const int n3 = in_sizes[1] / 27;
    const int n2 = in_sizes[3] / 27;
    const int n1 = in_sizes[5] / 27;

    const int* nbr4  = (const int*)d_in[0];
    const int* inv43 = (const int*)d_in[1];
    const int* nbr3  = (const int*)d_in[2];
    const int* inv32 = (const int*)d_in[3];
    const int* nbr2  = (const int*)d_in[4];
    const int* inv21 = (const int*)d_in[5];
    const int* nbr1  = (const int*)d_in[6];
    const float* x   = (const float*)d_in[7];

    const float *W[8], *g[8], *b[8];
    for (int i = 0; i < 8; ++i) {
        W[i] = (const float*)d_in[8 + 3 * i];
        g[i] = (const float*)d_in[9 + 3 * i];
        b[i] = (const float*)d_in[10 + 3 * i];
    }

    // workspace layout
    size_t NP = ((size_t)n1 + 127) & ~(size_t)127;  // n1 is the max point count
    char* ws = (char*)d_ws;
    float* yBuf = (float*)ws;                    ws += NP * 64 * sizeof(float);
    unsigned short* f0 = (unsigned short*)ws;    ws += NP * 64 * 2;
    unsigned short* f1 = (unsigned short*)ws;    ws += NP * 64 * 2;
    unsigned short* wpack = (unsigned short*)ws; ws += (size_t)4 * 54 * 512 * 2;
    unsigned short* zpage = (unsigned short*)ws; ws += 256;
    float* stats = (float*)ws;                   // 256 floats

    // zero page for out-of-range / masked gathers
    hipMemsetAsync(zpage, 0, 256, stream);

    // Convert input features to bf16
    long nx = (long)n4 * 64;
    f32_to_bf16_kernel<<<(int)((nx + 255) / 256), 256, 0, stream>>>(x, f0, nx);

    // m4: 64 -> 64, N = n4
    run_block<4, 6>(f0, nbr4, W[0], g[0], b[0], n4, 64, f1, nullptr, true, false,
                    yBuf, wpack, zpage, stats, stream);
    // i4: 64 -> 64, N = n3
    run_block<4, 6>(f1, inv43, W[1], g[1], b[1], n3, 64, f0, nullptr, true, false,
                    yBuf, wpack, zpage, stats, stream);
    // m3: 64 -> 64, N = n3
    run_block<4, 6>(f0, nbr3, W[2], g[2], b[2], n3, 64, f1, nullptr, true, false,
                    yBuf, wpack, zpage, stats, stream);
    // i3: 64 -> 32, N = n2
    run_block<2, 6>(f1, inv32, W[3], g[3], b[3], n2, 32, f0, nullptr, true, false,
                    yBuf, wpack, zpage, stats, stream);
    // m2: 32 -> 32, N = n2
    run_block<2, 5>(f0, nbr2, W[4], g[4], b[4], n2, 32, f1, nullptr, true, false,
                    yBuf, wpack, zpage, stats, stream);
    // i2: 32 -> 16, N = n1
    run_block<1, 5>(f1, inv21, W[5], g[5], b[5], n1, 16, f0, nullptr, true, false,
                    yBuf, wpack, zpage, stats, stream);
    // m1: 16 -> 16, N = n1
    run_block<1, 4>(f0, nbr1, W[6], g[6], b[6], n1, 16, f1, nullptr, true, false,
                    yBuf, wpack, zpage, stats, stream);
    // c5: 16 -> 3, N = n1, no relu, fp32 output
    run_block<1, 4>(f1, nbr1, W[7], g[7], b[7], n1, 3, nullptr, (float*)d_out,
                    false, true, yBuf, wpack, zpage, stats, stream);
}
